// ScaledDotProductAttention_78941498900860
// MI455X (gfx1250) — matrix operand order repaired and checked by reading, not compiled, not run on hardware
//
#include <hip/hip_runtime.h>
#include <stdint.h>
#include <stddef.h>

// Problem sizes (fixed by the reference)
constexpr int kB = 16;
constexpr int kS = 2048;
constexpr int kD = 128;
constexpr float kScale = 0.08838834764831845f; // 1/sqrt(128)

typedef __attribute__((ext_vector_type(16))) __bf16 v16bf;
typedef __attribute__((ext_vector_type(8)))  float  v8f;

// LDS strides (elements), padded to avoid systematic bank conflicts
constexpr int kLdK = 132;   // K tile rows: 32 x 128 bf16, stride 132
constexpr int kLdV = 34;    // V^T rows: 128 x 32 bf16, stride 34

// Load 16 contiguous bf16 (32B) from LDS via 8 dword reads (4B-aligned).
__device__ __forceinline__ v16bf lds_load16bf(const __bf16* p) {
  union { v16bf v; unsigned int u[8]; } r;
  const unsigned int* pu = (const unsigned int*)p;
#pragma unroll
  for (int i = 0; i < 8; ++i) r.u[i] = pu[i];
  return r.v;
}

// Per-thread register tile: 16 consecutive floats of one K/V row (4 x float4).
struct TileRegs { float4 f[4]; };

__device__ __forceinline__ void load_tile_regs(TileRegs& r, const float* __restrict__ base,
                                               int k0, int t) {
  const int key = t >> 3;            // 0..31
  const int d0  = (t & 7) << 4;      // 0..112 step 16
  const float4* src = (const float4*)(base + (size_t)(k0 + key) * kD + d0);
#pragma unroll
  for (int q4 = 0; q4 < 4; ++q4) r.f[q4] = src[q4];
}

// K tile regs -> bf16 LDS row major [key][d]
__device__ __forceinline__ void store_k_lds(const TileRegs& r, __bf16* ldsK, int t) {
  const int key = t >> 3;
  const int d0  = (t & 7) << 4;
  __bf16* dst = ldsK + key * kLdK + d0;
#pragma unroll
  for (int q4 = 0; q4 < 4; ++q4) {
    dst[q4 * 4 + 0] = (__bf16)r.f[q4].x;
    dst[q4 * 4 + 1] = (__bf16)r.f[q4].y;
    dst[q4 * 4 + 2] = (__bf16)r.f[q4].z;
    dst[q4 * 4 + 3] = (__bf16)r.f[q4].w;
  }
}

// V tile regs -> bf16 LDS transposed [d][key]
__device__ __forceinline__ void store_vt_lds(const TileRegs& r, __bf16* ldsVt, int t) {
  const int key = t >> 3;
  const int d0  = (t & 7) << 4;
#pragma unroll
  for (int q4 = 0; q4 < 4; ++q4) {
    ldsVt[(d0 + q4 * 4 + 0) * kLdV + key] = (__bf16)r.f[q4].x;
    ldsVt[(d0 + q4 * 4 + 1) * kLdV + key] = (__bf16)r.f[q4].y;
    ldsVt[(d0 + q4 * 4 + 2) * kLdV + key] = (__bf16)r.f[q4].z;
    ldsVt[(d0 + q4 * 4 + 3) * kLdV + key] = (__bf16)r.f[q4].w;
  }
}

#define WMMA_BF16(A, Bm, C) \
  __builtin_amdgcn_wmma_f32_16x16x32_bf16(false, (A), false, (Bm), (short)0, (C), false, false)

__global__ __launch_bounds__(256)
void sdpa_fused_kernel(const float* __restrict__ Q, const float* __restrict__ K,
                       const float* __restrict__ V, const int* __restrict__ mask,
                       float* __restrict__ outp, float* __restrict__ attnp) {
  __shared__ __align__(16) __bf16 ldsK[32 * kLdK];
  __shared__ __align__(16) __bf16 ldsVt[kD * kLdV];
  __shared__ __align__(16) __bf16 ldsP[8 * 16 * 32];   // per-wave P-tile scratch

  const int t    = threadIdx.x;
  const int wave = t >> 5;
  const int lane = t & 31;
  const int n    = lane & 15;   // B/C column within tile
  const int hf   = lane >> 4;   // half-wave select

  const int b     = blockIdx.x >> 4;            // 16 q-blocks per batch
  const int qbase = (blockIdx.x & 15) * 128 + wave * 16;

  const float* Qb = Q + (size_t)b * kS * kD;
  const float* Kb = K + (size_t)b * kS * kD;
  const float* Vb = V + (size_t)b * kS * kD;
  const int*   mb = mask + (size_t)b * kS;

  const v8f vzero = {0.f, 0.f, 0.f, 0.f, 0.f, 0.f, 0.f, 0.f};

  // ---- Build Q A-fragments (16x32 bf16, 4 chunks over D=128); fold in 1/sqrt(D) ----
  v16bf qf[4];
  {
    const float* Qrow = Qb + (size_t)(qbase + n) * kD;   // A rows map to lane%16
#pragma unroll
    for (int cc = 0; cc < 4; ++cc) {
      union { v16bf v; __bf16 e[16]; } tmp;
#pragma unroll
      for (int i = 0; i < 16; ++i) {
        const int pi = i >> 1, j = i & 1;
        const int kk = pi * 2 + 16 * hf + j;
        tmp.e[i] = (__bf16)(Qrow[cc * 32 + kk] * kScale);
      }
      qf[cc] = tmp.v;
    }
  }

  // Per-lane online softmax state (each lane owns key columns {n, n+16} mod 32).
  float mrow[8], lrow[8];
#pragma unroll
  for (int r = 0; r < 8; ++r) { mrow[r] = -INFINITY; lrow[r] = 0.f; }

  TileRegs kreg;
  load_tile_regs(kreg, Kb, 0, t);

  // ================= Pass 1: per-lane online row max / exp-sum =================
  for (int k0 = 0; k0 < kS; k0 += 32) {
    __syncthreads();                    // all readers of previous tile done
    store_k_lds(kreg, ldsK, t);
    __syncthreads();
    if (k0 + 32 < kS) load_tile_regs(kreg, Kb, k0 + 32, t);   // prefetch next tile

    const int mk0 = mb[k0 + n];
    const int mk1 = mb[k0 + 16 + n];

    v8f c0 = vzero, c1 = vzero;
#pragma unroll
    for (int cc = 0; cc < 4; ++cc) {
      v16bf bk0 = lds_load16bf(ldsK + n * kLdK + cc * 32 + hf * 16);
      v16bf bk1 = lds_load16bf(ldsK + (16 + n) * kLdK + cc * 32 + hf * 16);
      c0 = WMMA_BF16(qf[cc], bk0, c0);
      c1 = WMMA_BF16(qf[cc], bk1, c1);
    }
#pragma unroll
    for (int r = 0; r < 8; ++r) {
      float s0 = mk0 ? c0[r] : -INFINITY;
      float s1 = mk1 ? c1[r] : -INFINITY;
      float nm = fmaxf(mrow[r], fmaxf(s0, s1));
      float corr = __expf(mrow[r] - nm);
      lrow[r] = lrow[r] * corr + __expf(s0 - nm) + __expf(s1 - nm);
      mrow[r] = nm;
    }
  }

  // Merge (m, l) pairs across the 16 lanes of each half (4-step butterfly).
#pragma unroll
  for (int r = 0; r < 8; ++r) {
#pragma unroll
    for (int mstep = 1; mstep < 16; mstep <<= 1) {
      float mo = __shfl_xor(mrow[r], mstep, 32);
      float lo = __shfl_xor(lrow[r], mstep, 32);
      float nm = fmaxf(mrow[r], mo);
      lrow[r] = lrow[r] * __expf(mrow[r] - nm) + lo * __expf(mo - nm);
      mrow[r] = nm;
    }
  }

  float linv[8];
#pragma unroll
  for (int r = 0; r < 8; ++r) linv[r] = 1.0f / lrow[r];

  v8f oacc[8];
#pragma unroll
  for (int nn = 0; nn < 8; ++nn) oacc[nn] = vzero;

  __bf16* myP = ldsP + wave * (16 * 32);

  TileRegs vreg;
  load_tile_regs(kreg, Kb, 0, t);
  load_tile_regs(vreg, Vb, 0, t);

  // ===== Pass 2: recompute scores, write attn, accumulate out = P.V =====
  for (int k0 = 0; k0 < kS; k0 += 32) {
    __syncthreads();
    store_k_lds(kreg, ldsK, t);
    store_vt_lds(vreg, ldsVt, t);
    __syncthreads();
    if (k0 + 32 < kS) {                 // prefetch next tiles
      load_tile_regs(kreg, Kb, k0 + 32, t);
      load_tile_regs(vreg, Vb, k0 + 32, t);
    }

    const int mk0 = mb[k0 + n];
    const int mk1 = mb[k0 + 16 + n];

    v8f c0 = vzero, c1 = vzero;
#pragma unroll
    for (int cc = 0; cc < 4; ++cc) {
      v16bf bk0 = lds_load16bf(ldsK + n * kLdK + cc * 32 + hf * 16);
      v16bf bk1 = lds_load16bf(ldsK + (16 + n) * kLdK + cc * 32 + hf * 16);
      c0 = WMMA_BF16(qf[cc], bk0, c0);
      c1 = WMMA_BF16(qf[cc], bk1, c1);
    }

    // Normalize, stream attn to HBM, park bf16 P tile in LDS for layout swap.
#pragma unroll
    for (int r = 0; r < 8; ++r) {
      const int qrow = r + 8 * hf;                 // C-layout row
      float p0 = mk0 ? __expf(c0[r] - mrow[r]) * linv[r] : 0.f;
      float p1 = mk1 ? __expf(c1[r] - mrow[r]) * linv[r] : 0.f;
      const size_t arow = ((size_t)b * kS + (qbase + qrow)) * (size_t)kS + k0;
      attnp[arow + n]      = p0;
      attnp[arow + 16 + n] = p1;
      myP[qrow * 32 + n]      = (__bf16)p0;
      myP[qrow * 32 + 16 + n] = (__bf16)p1;
    }

    // Rebuild P as an A-fragment (16x32) from LDS (intra-wave, no barrier needed).
    v16bf pf;
    {
      union { v16bf v; unsigned int u[8]; } tmp;
#pragma unroll
      for (int pi = 0; pi < 8; ++pi) {
        const int kk = pi * 2 + 16 * hf;
        tmp.u[pi] = *(const unsigned int*)(myP + n * 32 + kk);
      }
      pf = tmp.v;
    }

    // out tile: 8 WMMAs over the 128-wide output dim, B from transposed V.
#pragma unroll
    for (int nn = 0; nn < 8; ++nn) {
      v16bf bv = lds_load16bf(ldsVt + (nn * 16 + n) * kLdV + hf * 16);
      oacc[nn] = WMMA_BF16(pf, bv, oacc[nn]);
    }
  }

  // ---- Write out [16 x 128] tile ----
#pragma unroll
  for (int nn = 0; nn < 8; ++nn) {
#pragma unroll
    for (int r = 0; r < 8; ++r) {
      const int qrow = qbase + r + 8 * hf;
      outp[((size_t)b * kS + qrow) * kD + nn * 16 + n] = oacc[nn][r];
    }
  }
}

extern "C" void kernel_launch(void* const* d_in, const int* in_sizes, int n_in,
                              void* d_out, int out_size, void* d_ws, size_t ws_size,
                              hipStream_t stream) {
  (void)in_sizes; (void)n_in; (void)out_size; (void)d_ws; (void)ws_size;
  const float* Q    = (const float*)d_in[0];
  const float* K    = (const float*)d_in[1];
  const float* V    = (const float*)d_in[2];
  const int*   mask = (const int*)d_in[3];

  float* outp  = (float*)d_out;                        // [B,S,D]
  float* attnp = outp + (size_t)kB * kS * kD;          // [B,S,S]

  dim3 grid(kB * (kS / 128));   // 256 blocks, 8 waves each
  dim3 block(256);
  sdpa_fused_kernel<<<grid, block, 0, stream>>>(Q, K, V, mask, outp, attnp);
}
